// SwinBlockWrapper_59811714564537
// MI455X (gfx1250) — matrix operand order repaired and checked by reading, not compiled, not run on hardware
//
#include <hip/hip_runtime.h>
#include <hip/hip_bf16.h>
#include <math.h>

// ---------------- problem constants ----------------
#define BSZ   64
#define HH    56
#define WW_   56
#define CC    128
#define WS_   7
#define SHIFT_ 3
#define NH_   4
#define HD_   32
#define NTOK  49          // tokens per window
#define NWIN  64          // windows per image
#define HID_  512
#define TOKENS (BSZ * HH * WW_)   // 200704

typedef __attribute__((ext_vector_type(16))) _Float16 v16h;
typedef __attribute__((ext_vector_type(8)))  _Float16 v8h;
typedef __attribute__((ext_vector_type(8)))  float    v8f;

// ---------------- WMMA helpers (gfx1250, wave32) ----------------
__device__ __forceinline__ v8f wmma16(v16h a, v16h b, v8f c) {
  // D = A(16x32 f16) * B(32x16 f16) + C(16x16 f32)
  return __builtin_amdgcn_wmma_f32_16x16x32_f16(false, a, false, b, (short)0, c,
                                                false, false);
}

// A fragment: 16x32 f16 from row-major src[row][col], leading dim ld (elements).
// ISA layout: lanes 0-15 -> M=lane, K in {0..7,16..23}; lanes 16-31 -> M=lane-16,
// K in {8..15,24..31}. Loads are 16B-aligned for all our ld/kb combinations.
__device__ __forceinline__ v16h frag_A(const _Float16* __restrict__ src, int ld) {
  const int l  = threadIdx.x & 31;
  const int m  = l & 15;
  const int kb = (l & 16) ? 8 : 0;
  const _Float16* row = src + m * ld;
  const v8h lo = *(const v8h*)(row + kb);
  const v8h hi = *(const v8h*)(row + 16 + kb);
  v16h a;
#pragma unroll
  for (int j = 0; j < 8; ++j) { a[j] = lo[j]; a[8 + j] = hi[j]; }
  return a;
}

// B fragment: 32x16 f16 where B[k][n] is stored n-major: src[n][k], leading dim ld.
// ISA layout: lanes 0-15 -> N=lane, K in {0..7,16..23}; lanes 16-31 -> N=lane-16,
// K in {8..15,24..31}.
__device__ __forceinline__ v16h frag_B(const _Float16* __restrict__ src, int ld) {
  const int l  = threadIdx.x & 31;
  const int n  = l & 15;
  const int kb = (l & 16) ? 8 : 0;
  const _Float16* row = src + n * ld + kb;
  const v8h lo = *(const v8h*)(row);
  const v8h hi = *(const v8h*)(row + 16);
  v16h b;
#pragma unroll
  for (int j = 0; j < 8; ++j) { b[j] = lo[j]; b[8 + j] = hi[j]; }
  return b;
}

// ---------------- prep: weights fp32 -> f16, per-head bias gather ----------------
#define SZ_QKVW (3 * CC * CC)      // 49152
#define SZ_PROJW (CC * CC)         // 16384
#define SZ_FC1W (HID_ * CC)        // 65536
#define SZ_FC2W (CC * HID_)        // 65536
#define SZ_BIAS (NH_ * NTOK * NTOK) // 9604
#define PREP_TOTAL (SZ_QKVW + SZ_PROJW + SZ_FC1W + SZ_FC2W + SZ_BIAS)

__global__ void prep_kernel(const float* __restrict__ qkv_w,
                            const float* __restrict__ proj_w,
                            const float* __restrict__ fc1_w,
                            const float* __restrict__ fc2_w,
                            const float* __restrict__ bias_table,
                            const int*   __restrict__ rel_index,
                            _Float16* __restrict__ wq, _Float16* __restrict__ wp,
                            _Float16* __restrict__ w1, _Float16* __restrict__ w2,
                            float* __restrict__ biasNH) {
  int i = blockIdx.x * blockDim.x + threadIdx.x;
  if (i < SZ_QKVW) { wq[i] = (_Float16)qkv_w[i]; return; }
  i -= SZ_QKVW;
  if (i < SZ_PROJW) { wp[i] = (_Float16)proj_w[i]; return; }
  i -= SZ_PROJW;
  if (i < SZ_FC1W) { w1[i] = (_Float16)fc1_w[i]; return; }
  i -= SZ_FC1W;
  if (i < SZ_FC2W) { w2[i] = (_Float16)fc2_w[i]; return; }
  i -= SZ_FC2W;
  if (i < SZ_BIAS) {
    const int h = i / (NTOK * NTOK), ij = i % (NTOK * NTOK);
    biasNH[h * (NTOK * NTOK) + ij] = bias_table[rel_index[ij] * NH_ + h];
  }
}

// ---------------- attention kernel: one block (8 waves) per window ----------------
// LDS map (bytes), 112 KB total -> 2 blocks / WGP (320 KB):
//  xa  [64][128] f16        @      0  (16384)
//  qs  [4][64][32] f16      @  16384  (16384)   (q pre-scaled by 1/sqrt(HD))
//  ks  [4][64][32] f16      @  32768  (16384)
//  vt  [4][32][64] f16      @  49152  (16384)   (v transposed: [chan][token])
//  ps  [4][64][64] f16      @  65536  (32768)   (logits, then softmax in place)
//  os  [64][128] f16        @  98304  (16384)
#define ATTN_LDS 114688

__global__ __launch_bounds__(256) void
attn_kernel(const float* __restrict__ x,
            const float* __restrict__ xn1,
            const _Float16* __restrict__ wqkv, const float* __restrict__ bqkv,
            const _Float16* __restrict__ wproj, const float* __restrict__ bproj,
            const float* __restrict__ biasNH,
            const float* __restrict__ mask,
            float* __restrict__ xattn) {
  extern __shared__ char smem[];
  _Float16* xa = (_Float16*)(smem);
  _Float16* qs = (_Float16*)(smem + 16384);
  _Float16* ks = (_Float16*)(smem + 32768);
  _Float16* vt = (_Float16*)(smem + 49152);
  _Float16* ps = (_Float16*)(smem + 65536);
  _Float16* os = (_Float16*)(smem + 98304);

  const int tid  = threadIdx.x;
  // scalarized wave id: tile loops, weight bases and q/k/v branch become SALU
  const int wvu  = __builtin_amdgcn_readfirstlane(tid >> 5);
  const int lane = tid & 31;
  const int nlan = lane & 15;
  const int mb   = (lane & 16) ? 8 : 0;

  const int blk = blockIdx.x;
  const int b   = blk >> 6;      // image
  const int wi  = blk & 63;      // window in image
  const int wh  = wi >> 3, ww = wi & 7;

  // ---- phase 0: shifted-window gather, fp32 -> f16 A tile (pad rows 49..63) ----
  for (int idx = tid; idx < 64 * 128; idx += 256) {
    const int r = idx >> 7, c = idx & 127;
    _Float16 val = (_Float16)0.0f;
    if (r < NTOK) {
      const int iy = r / WS_, ix = r % WS_;
      const int gh = (wh * WS_ + iy + SHIFT_) % HH;
      const int gw = (ww * WS_ + ix + SHIFT_) % WW_;
      val = (_Float16)xn1[((size_t)b * (HH * WW_) + gh * WW_ + gw) * CC + c];
    }
    xa[idx] = val;
  }
  __syncthreads();

  // ---- phase 1: QKV = xa(64x128) @ wqkv^T(128x384) + b ----
  const float qscale = 0.17677669529663689f;  // HD^-0.5
  for (int t = wvu; t < 96; t += 8) {         // 4 m-tiles x 24 n-tiles
    const int mt = t & 3, nt = t >> 2;
    v8f acc = {};
#pragma unroll
    for (int kk = 0; kk < 4; ++kk) {
      const v16h a  = frag_A(xa + mt * 16 * 128 + kk * 32, 128);
      const v16h bf = frag_B(wqkv + (nt * 16) * 128 + kk * 32, 128);
      acc = wmma16(a, bf, acc);
    }
    // scalar (tile-uniform) decode: part / head from nt only
    const int part = nt >> 3;                 // 0=q,1=k,2=v
    const int h    = (nt >> 1) & 3;
    const int ch   = ((nt & 1) << 4) + nlan;  // channel within head
    const float bias = bqkv[nt * 16 + nlan];
    if (part == 0) {
#pragma unroll
      for (int r = 0; r < 8; ++r) {
        const int m = mt * 16 + mb + r;
        qs[(h * 64 + m) * 32 + ch] = (_Float16)((acc[r] + bias) * qscale);
      }
    } else if (part == 1) {
#pragma unroll
      for (int r = 0; r < 8; ++r) {
        const int m = mt * 16 + mb + r;
        ks[(h * 64 + m) * 32 + ch] = (_Float16)(acc[r] + bias);
      }
    } else {
#pragma unroll
      for (int r = 0; r < 8; ++r) {
        const int m = mt * 16 + mb + r;
        vt[(h * 32 + ch) * 64 + m] = (_Float16)(acc[r] + bias);
      }
    }
  }
  __syncthreads();

  // ---- phase 2: S = q @ k^T  (wave (h, mhalf); logits stored f16 in ps) ----
  {
    const int h = wvu & 3;
    const int mhalf = wvu >> 2;
    for (int t = 0; t < 8; ++t) {             // 2 m-tiles x 4 n-tiles, K=32
      const int mt = (mhalf << 1) + (t & 1), nt = t >> 1;
      const v16h a  = frag_A(qs + h * 2048 + mt * 16 * 32, 32);
      const v16h bf = frag_B(ks + h * 2048 + nt * 16 * 32, 32);
      v8f acc = {};
      acc = wmma16(a, bf, acc);
      const int nglob = nt * 16 + nlan;
#pragma unroll
      for (int r = 0; r < 8; ++r) {
        const int m = mt * 16 + mb + r;
        float v;
        if (m >= NTOK)           v = 0.0f;
        else if (nglob >= NTOK)  v = -30000.0f;   // f16-safe -inf surrogate
        else v = acc[r] + biasNH[h * (NTOK * NTOK) + m * NTOK + nglob]
                        + mask[((size_t)wi * NTOK + m) * NTOK + nglob];
        ps[(h * 64 + m) * 64 + nglob] = (_Float16)v;
      }
    }
  }
  __syncthreads();

  // ---- phase 3: softmax in place over valid 49 columns (one row per thread) ----
  {
    const int row = tid;                      // 256 threads, 256 rows
    const int m = row & 63;
    _Float16* prow = ps + row * 64;
    if (m >= NTOK) {
      for (int n = 0; n < 64; ++n) prow[n] = (_Float16)0.0f;
    } else {
      float mx = -1e30f;
      for (int n = 0; n < NTOK; ++n) mx = fmaxf(mx, (float)prow[n]);
      float sum = 0.0f;
      for (int n = 0; n < NTOK; ++n) {
        const float e = expf((float)prow[n] - mx);
        sum += e;
        prow[n] = (_Float16)e;                // unnormalized exp
      }
      const float inv = 1.0f / sum;
      for (int n = 0; n < NTOK; ++n) prow[n] = (_Float16)((float)prow[n] * inv);
      for (int n = NTOK; n < 64; ++n) prow[n] = (_Float16)0.0f;
    }
  }
  __syncthreads();

  // ---- phase 4: O = P @ V  (wave (h, mhalf)) ----
  {
    const int h = wvu & 3;
    const int mhalf = wvu >> 2;
    for (int t = 0; t < 4; ++t) {             // 2 m-tiles x 2 n-tiles, K=64
      const int mt = (mhalf << 1) + (t & 1), nt = t >> 1;
      v8f acc = {};
#pragma unroll
      for (int kk = 0; kk < 2; ++kk) {
        const v16h a  = frag_A(ps + h * 4096 + mt * 16 * 64 + kk * 32, 64);
        const v16h bf = frag_B(vt + h * 2048 + nt * 16 * 64 + kk * 32, 64);
        acc = wmma16(a, bf, acc);
      }
      const int nglob = nt * 16 + nlan;
#pragma unroll
      for (int r = 0; r < 8; ++r) {
        const int m = mt * 16 + mb + r;
        os[m * 128 + h * 32 + nglob] = (_Float16)acc[r];
      }
    }
  }
  __syncthreads();

  // ---- phase 5: proj + residual, scatter with reverse shift ----
  for (int t = wvu; t < 32; t += 8) {         // 4 m-tiles x 8 n-tiles
    const int mt = t & 3, nt = t >> 2;
    v8f acc = {};
#pragma unroll
    for (int kk = 0; kk < 4; ++kk) {
      const v16h a  = frag_A(os + mt * 16 * 128 + kk * 32, 128);
      const v16h bf = frag_B(wproj + (nt * 16) * 128 + kk * 32, 128);
      acc = wmma16(a, bf, acc);
    }
    const int nglob = nt * 16 + nlan;
    const float bias = bproj[nglob];
#pragma unroll
    for (int r = 0; r < 8; ++r) {
      const int m = mt * 16 + mb + r;
      if (m < NTOK) {
        const int iy = m / WS_, ix = m % WS_;
        const int gh = (wh * WS_ + iy + SHIFT_) % HH;
        const int gw = (ww * WS_ + ix + SHIFT_) % WW_;
        const size_t o = ((size_t)b * (HH * WW_) + gh * WW_ + gw) * CC + nglob;
        xattn[o] = acc[r] + bias + x[o];
      }
    }
  }
}

// ---------------- fused LN + FC1(GELU) + FC2 + residual ----------------
// LDS: a_lds [64][128] f16 @ 0 (16384); h_lds [64][512] f16 @ 16384 (65536)
#define MLP_LDS 81920

__global__ __launch_bounds__(256) void
mlp_kernel(const float* __restrict__ xattn,
           const float* __restrict__ n2w, const float* __restrict__ n2b,
           const _Float16* __restrict__ w1, const float* __restrict__ b1,
           const _Float16* __restrict__ w2, const float* __restrict__ b2,
           float* __restrict__ out) {
  extern __shared__ char smem[];
  _Float16* a_lds = (_Float16*)(smem);
  _Float16* h_lds = (_Float16*)(smem + 16384);

  const int tid  = threadIdx.x;
  const int wvu  = __builtin_amdgcn_readfirstlane(tid >> 5);
  const int lane = tid & 31;
  const int nlan = lane & 15;
  const int mb   = (lane & 16) ? 8 : 0;
  const size_t row0 = (size_t)blockIdx.x * 64;

  // ---- LayerNorm, one row per thread (threads 0..63) ----
  if (tid < 64) {
    const float* xr = xattn + (row0 + tid) * CC;
    float mu = 0.0f;
    for (int c = 0; c < CC; ++c) mu += xr[c];
    mu *= (1.0f / CC);
    float var = 0.0f;
    for (int c = 0; c < CC; ++c) { const float d = xr[c] - mu; var += d * d; }
    var *= (1.0f / CC);
    const float rs = rsqrtf(var + 1e-5f);
    for (int c = 0; c < CC; ++c)
      a_lds[tid * CC + c] = (_Float16)((xr[c] - mu) * rs * n2w[c] + n2b[c]);
  }
  __syncthreads();

  // ---- FC1 + exact GELU: (64x128) @ (128x512) ----
  for (int t = wvu; t < 128; t += 8) {        // 4 m-tiles x 32 n-tiles
    const int mt = t & 3, nt = t >> 2;
    v8f acc = {};
#pragma unroll
    for (int kk = 0; kk < 4; ++kk) {
      const v16h a  = frag_A(a_lds + mt * 16 * 128 + kk * 32, 128);
      const v16h bf = frag_B(w1 + (nt * 16) * 128 + kk * 32, 128);
      acc = wmma16(a, bf, acc);
    }
    const int nglob = nt * 16 + nlan;
    const float bias = b1[nglob];
#pragma unroll
    for (int r = 0; r < 8; ++r) {
      float v = acc[r] + bias;
      v = 0.5f * v * (1.0f + erff(v * 0.70710678118654752f));
      h_lds[(mt * 16 + mb + r) * HID_ + nglob] = (_Float16)v;
    }
  }
  __syncthreads();

  // ---- FC2 + residual: (64x512) @ (512x128) ----
  for (int t = wvu; t < 32; t += 8) {         // 4 m-tiles x 8 n-tiles
    const int mt = t & 3, nt = t >> 2;
    v8f acc = {};
#pragma unroll
    for (int kk = 0; kk < 16; ++kk) {
      const v16h a  = frag_A(h_lds + mt * 16 * HID_ + kk * 32, HID_);
      const v16h bf = frag_B(w2 + (nt * 16) * HID_ + kk * 32, HID_);
      acc = wmma16(a, bf, acc);
    }
    const int nglob = nt * 16 + nlan;
    const float bias = b2[nglob];
#pragma unroll
    for (int r = 0; r < 8; ++r) {
      const size_t row = row0 + mt * 16 + mb + r;
      out[row * CC + nglob] = acc[r] + bias + xattn[row * CC + nglob];
    }
  }
}

// ---------------- workspace layout (bytes) ----------------
#define WS_QKVW  0u
#define WS_PROJW 98304u
#define WS_FC1W  131072u
#define WS_FC2W  262144u
#define WS_BIAS  393216u
#define WS_XATTN 458752u   // TOKENS*CC*4 = 102,760,448 bytes

extern "C" void kernel_launch(void* const* d_in, const int* in_sizes, int n_in,
                              void* d_out, int out_size, void* d_ws, size_t ws_size,
                              hipStream_t stream) {
  (void)in_sizes; (void)n_in; (void)out_size; (void)ws_size;
  const float* x      = (const float*)d_in[0];
  const float* xn1    = (const float*)d_in[1];
  const float* qkv_w  = (const float*)d_in[2];
  const float* qkv_b  = (const float*)d_in[3];
  const float* reltab = (const float*)d_in[4];
  const float* proj_w = (const float*)d_in[5];
  const float* proj_b = (const float*)d_in[6];
  const float* n2w    = (const float*)d_in[7];
  const float* n2b    = (const float*)d_in[8];
  const float* fc1_w  = (const float*)d_in[9];
  const float* fc1_b  = (const float*)d_in[10];
  const float* fc2_w  = (const float*)d_in[11];
  const float* fc2_b  = (const float*)d_in[12];
  const float* amask  = (const float*)d_in[13];
  const int*   relidx = (const int*)d_in[14];

  char* ws = (char*)d_ws;
  _Float16* wq = (_Float16*)(ws + WS_QKVW);
  _Float16* wp = (_Float16*)(ws + WS_PROJW);
  _Float16* w1 = (_Float16*)(ws + WS_FC1W);
  _Float16* w2 = (_Float16*)(ws + WS_FC2W);
  float* biasNH = (float*)(ws + WS_BIAS);
  float* xattn  = (float*)(ws + WS_XATTN);
  float* out = (float*)d_out;

  prep_kernel<<<(PREP_TOTAL + 255) / 256, 256, 0, stream>>>(
      qkv_w, proj_w, fc1_w, fc2_w, reltab, relidx, wq, wp, w1, w2, biasNH);

  attn_kernel<<<BSZ * NWIN, 256, ATTN_LDS, stream>>>(
      x, xn1, wq, qkv_b, wp, proj_b, biasNH, amask, xattn);

  mlp_kernel<<<TOKENS / 64, 256, MLP_LDS, stream>>>(
      xattn, n2w, n2b, w1, fc1_b, w2, fc2_b, out);
}
